// HeteroFloodGNN_54382875902387
// MI455X (gfx1250) — compile-verified, matrix operand order checked by reading, not run on hardware
//
#include <hip/hip_runtime.h>
#include <stddef.h>
#include <stdint.h>

// ---------------------------------------------------------------------------
// HeteroFloodGNN inference for MI455X (gfx1250, wave32, WMMA + TDM).
//
// - Node features stored bf16 (halves gather/GEMM-A traffic; WMMA-native),
//   fp32 masters kept for residual+LayerNorm numerics.
// - All [N,128]x[128,128] linears go through one WMMA kernel:
//   * W (32KB bf16) DMA'd into LDS once per block via TENSOR_LOAD_TO_LDS
//     (Tensor Data Mover), waited with s_wait_tensorcnt, then barrier.
//   * B fragments read from LDS (ds_load_b128), held in VGPRs and reused
//     across 4 m-tiles (64 rows) per block.
//   * K=128 via 4 chained v_wmma_f32_16x16x32_bf16, fp32 accumulate.
//   * SAGE mean (agg/count) fused into the A-fragment load.
// - Edge aggregation: wave-per-edge gather + global_atomic_add_f32 scatter.
//
// Param flattening assumption: setup_inputs() leaves in dict insertion
// order, depth first:
//   d_in[0..5]: x1d, x2d, ei_pipe, ei_surf, ei_c12, ei_c21
//   then params: enc1d[0].{W,b}, enc1d[1].{W,b}, enc2d[0..1].{W,b},
//                dec1d[0..1].{W,b}, dec2d[0..1].{W,b},
//                layers[l]: pipe{Wl,bl,Wr}, surf{Wl,bl,Wr},
//                           c12{Wl,bl,Wr}, c21{Wl,bl,Wr},
//                           ln1d{g,b}, ln2d{g,b}
// ---------------------------------------------------------------------------

typedef __bf16 bf16;
typedef __attribute__((ext_vector_type(16))) bf16  v16bf;
typedef __attribute__((ext_vector_type(8)))  bf16  v8bf;
typedef __attribute__((ext_vector_type(4)))  bf16  v4bf;
typedef __attribute__((ext_vector_type(8)))  float v8f;
typedef __attribute__((ext_vector_type(4)))  unsigned int v4u;
typedef __attribute__((ext_vector_type(8)))  int   v8i;
typedef __attribute__((ext_vector_type(4)))  int   v4i;

#define GF_ACC  1   // accumulate into C (read-modify-write)
#define GF_GELU 2   // apply exact GELU in epilogue

__device__ __forceinline__ float gelu_f(float x) {
  return 0.5f * x * (1.0f + erff(x * 0.70710678118654752f));
}

// Load one 16x32 bf16 WMMA fragment (A or B^T) from a row-major 128-wide row.
// ISA 7.12.2: lanes 0-15 hold K = k0+[0..7] / k0+16+[0..7];
// lanes 16-31 hold K = k0+8+[0..7] / k0+24+[0..7].
__device__ __forceinline__ v16bf frag_ld(const bf16* __restrict__ row, int k0, int kb) {
  v8bf lo = *(const v8bf*)(row + k0 + kb);
  v8bf hi = *(const v8bf*)(row + k0 + 16 + kb);
  v16bf r;
#pragma unroll
  for (int i = 0; i < 8; ++i) { r[i] = lo[i]; r[8 + i] = hi[i]; }
  return r;
}

// A fragment from fp32 edge-aggregate scaled by 1/count (SAGE mean fused).
__device__ __forceinline__ v16bf frag_ld_mean(const float* __restrict__ row,
                                              float inv, int k0, int kb) {
  v16bf r;
#pragma unroll
  for (int i = 0; i < 8; ++i) {
    r[i]     = (bf16)(row[k0 + kb + i] * inv);
    r[8 + i] = (bf16)(row[k0 + 16 + kb + i] * inv);
  }
  return r;
}

// TDM: DMA the 128x128 bf16 weight matrix (32KB) global -> LDS.
// D# per ISA ch.8: 2-D tensor, data_size=2B, tensor_dim0/1=128, tile 128x128,
// dim0 stride=128, group0.type=2 ("image"), groups 2/3 disabled (zeros).
// Returns false if the builtin is unavailable (caller falls back).
__device__ __forceinline__ bool tdm_load_w(const bf16* gW, unsigned ldsOff) {
#if __has_builtin(__builtin_amdgcn_tensor_load_to_lds)
  unsigned long long ga = (unsigned long long)(size_t)gW;
  v4u g0 = { 1u,                                    // count=1, user mode
             ldsOff,                                // lds_addr (bytes)
             (unsigned)(ga & 0xffffffffu),          // global_addr[31:0]
             (unsigned)((ga >> 32) & 0x1ffffffu)    // global_addr[56:32]
               | 0x80000000u };                     // type=2
  v8i g1 = { (int)(1u << 16),                       // data_size=1 (2 bytes)
             (int)(128u << 16),                     // tensor_dim0[15:0]=128
             (int)(128u << 16),                     // tensor_dim1[15:0]=128
             (int)(128u << 16),                     // tile_dim0=128
             128,                                   // tile_dim1=128 (tile_dim2=0)
             128,                                   // tensor_dim0_stride=128
             0, 0 };                                // dim1 stride unused (2-D)
  v4i z4 = { 0, 0, 0, 0 };
#if defined(__clang_major__) && __clang_major__ >= 23
  v8i z8 = { 0, 0, 0, 0, 0, 0, 0, 0 };
  __builtin_amdgcn_tensor_load_to_lds(g0, g1, z4, z4, z8, 0);
#else
  __builtin_amdgcn_tensor_load_to_lds(g0, g1, z4, z4, 0);
#endif
  __builtin_amdgcn_s_wait_tensorcnt(0);
  return true;
#else
  (void)gW; (void)ldsOff;
  return false;
#endif
}

// C[M,128] (+)= A[M,128] @ W^T, W row-major [128(out),128(in)] bf16.
// Block = 256 threads (8 waves) covering 64 rows x 128 cols:
// wave -> n-tile (16 cols), loops over 4 m-tiles reusing B fragments in VGPRs.
__global__ void __launch_bounds__(256)
k_wmma_gemm(const bf16* __restrict__ Ab, const float* __restrict__ Af,
            const float* __restrict__ cnt, const bf16* __restrict__ W,
            const float* __restrict__ bias, float* __restrict__ C,
            bf16* __restrict__ Cb, int M, int flags) {
  __shared__ bf16 smW[128 * 128];
  const int lane = threadIdx.x & 31;
  const int wid  = threadIdx.x >> 5;      // n-tile 0..7

  // Stage W into LDS: one TDM DMA issued by wave 0 (EXEC-independent),
  // other waves just meet it at the barrier.
  if (wid == 0) {
    if (!tdm_load_w(W, (unsigned)(uintptr_t)(void*)smW)) {
      for (int i = lane; i < 128 * 16; i += 32)       // fallback: 8 bf16/thread
        ((v8bf*)smW)[i] = ((const v8bf*)W)[i];
    }
  }
  __syncthreads();

  const int n0   = wid * 16;
  const int half = lane >> 4;             // 0 | 1
  const int kb   = half * 8;
  const int nB   = n0 + (lane & 15);
  const int col  = n0 + (lane & 15);

  // B fragments from LDS (ds_load_b128), reused across all 4 m-tiles.
  const bf16* wrow = smW + (size_t)nB * 128;
  const v16bf b0 = frag_ld(wrow,  0, kb);
  const v16bf b1 = frag_ld(wrow, 32, kb);
  const v16bf b2 = frag_ld(wrow, 64, kb);
  const v16bf b3 = frag_ld(wrow, 96, kb);

  const float bv = bias ? bias[col] : 0.0f;

#pragma unroll
  for (int t = 0; t < 4; ++t) {
    const int m0 = blockIdx.x * 64 + t * 16;
    if (m0 >= M) break;                   // block-uniform, EXEC untouched
    int rA = m0 + (lane & 15);
    if (rA >= M) rA = M - 1;              // clamp (keep EXEC all-ones)

    v16bf a0, a1, a2, a3;
    if (Af) {
      const float* arow = Af + (size_t)rA * 128;
      const float inv = 1.0f / fmaxf(cnt[rA], 1.0f);
      a0 = frag_ld_mean(arow, inv,  0, kb);
      a1 = frag_ld_mean(arow, inv, 32, kb);
      a2 = frag_ld_mean(arow, inv, 64, kb);
      a3 = frag_ld_mean(arow, inv, 96, kb);
    } else {
      const bf16* arow = Ab + (size_t)rA * 128;
      a0 = frag_ld(arow,  0, kb);
      a1 = frag_ld(arow, 32, kb);
      a2 = frag_ld(arow, 64, kb);
      a3 = frag_ld(arow, 96, kb);
    }

    // C/D layout: lane N = col; VGPR v -> M = m0 + half*8 + v.
    const int rbase = m0 + half * 8;
    v8f acc = {};
    if (flags & GF_ACC) {
#pragma unroll
      for (int v = 0; v < 8; ++v) {
        int r = rbase + v; if (r >= M) r = M - 1;
        acc[v] = C[(size_t)r * 128 + col];
      }
    }
#pragma unroll
    for (int v = 0; v < 8; ++v) acc[v] += bv;

    acc = __builtin_amdgcn_wmma_f32_16x16x32_bf16(false, a0, false, b0, (short)0, acc, false, false);
    acc = __builtin_amdgcn_wmma_f32_16x16x32_bf16(false, a1, false, b1, (short)0, acc, false, false);
    acc = __builtin_amdgcn_wmma_f32_16x16x32_bf16(false, a2, false, b2, (short)0, acc, false, false);
    acc = __builtin_amdgcn_wmma_f32_16x16x32_bf16(false, a3, false, b3, (short)0, acc, false, false);

#pragma unroll
    for (int v = 0; v < 8; ++v) {
      const int r = rbase + v;
      if (r < M) {
        float y = acc[v];
        if (flags & GF_GELU) y = gelu_f(y);
        C[(size_t)r * 128 + col] = y;
        if (Cb) Cb[(size_t)r * 128 + col] = (bf16)y;
      }
    }
  }
}

// Edge gather + segment-sum scatter: wave per edge, lane carries 4 features.
__global__ void __launch_bounds__(256)
k_gather(const bf16* __restrict__ hs, const int* __restrict__ ei, int E,
         float* __restrict__ agg, float* __restrict__ cnt) {
  const int tid  = blockIdx.x * 256 + threadIdx.x;
  const int e    = tid >> 5;
  const int lane = tid & 31;
  if (e >= E) return;
  const int s = ei[e];
  const int d = ei[E + e];
  v4bf v = *(const v4bf*)(hs + (size_t)s * 128 + lane * 4);
  float* ap = agg + (size_t)d * 128 + lane * 4;
  atomicAdd(ap + 0, (float)v[0]);
  atomicAdd(ap + 1, (float)v[1]);
  atomicAdd(ap + 2, (float)v[2]);
  atomicAdd(ap + 3, (float)v[3]);
  if (lane == 0) atomicAdd(cnt + d, 1.0f);
}

// Residual + LayerNorm; wave per row; updates fp32 master and bf16 mirror.
__global__ void __launch_bounds__(256)
k_ln(float* __restrict__ h, const float* __restrict__ nw,
     const float* __restrict__ g, const float* __restrict__ b,
     bf16* __restrict__ hb, int N) {
  const int lane = threadIdx.x & 31;
  const int row  = blockIdx.x * 8 + (threadIdx.x >> 5);
  if (row >= N) return;
  const size_t base = (size_t)row * 128 + lane * 4;
  float x[4];
#pragma unroll
  for (int i = 0; i < 4; ++i) x[i] = h[base + i] + nw[base + i];
  float s = x[0] + x[1] + x[2] + x[3];
#pragma unroll
  for (int m = 16; m >= 1; m >>= 1) s += __shfl_xor(s, m, 32);
  const float mu = s * (1.0f / 128.0f);
  float q = 0.0f;
#pragma unroll
  for (int i = 0; i < 4; ++i) { const float dd = x[i] - mu; q += dd * dd; }
#pragma unroll
  for (int m = 16; m >= 1; m >>= 1) q += __shfl_xor(q, m, 32);
  const float rs = rsqrtf(q * (1.0f / 128.0f) + 1e-5f);
#pragma unroll
  for (int i = 0; i < 4; ++i) {
    const int c = lane * 4 + i;
    const float y = (x[i] - mu) * rs * g[c] + b[c];
    h[base + i]  = y;
    hb[base + i] = (bf16)y;
  }
}

// First encoder layer: tiny K (10/15) -> plain VALU dot, GELU, bf16 out.
__global__ void __launch_bounds__(256)
k_enc1(const float* __restrict__ x, const float* __restrict__ W,
       const float* __restrict__ b, bf16* __restrict__ out, int N, int K) {
  const int tid = blockIdx.x * 256 + threadIdx.x;
  const int row = tid >> 7;
  const int col = tid & 127;
  if (row >= N) return;
  const float* xr = x + (size_t)row * K;
  const float* wr = W + (size_t)col * K;
  float s = b[col];
  for (int k = 0; k < K; ++k) s += xr[k] * wr[k];
  out[(size_t)row * 128 + col] = (bf16)gelu_f(s);
}

// Decoder head: out[row] = dot(hid[row,:], W1) + b1. Wave per row.
__global__ void __launch_bounds__(256)
k_dec_out(const float* __restrict__ hid, const float* __restrict__ W1,
          const float* __restrict__ b1, float* __restrict__ out, int N) {
  const int lane = threadIdx.x & 31;
  const int row  = blockIdx.x * 8 + (threadIdx.x >> 5);
  if (row >= N) return;
  const float* hr = hid + (size_t)row * 128 + lane * 4;
  const float* wr = W1 + lane * 4;
  float s = hr[0] * wr[0] + hr[1] * wr[1] + hr[2] * wr[2] + hr[3] * wr[3];
#pragma unroll
  for (int m = 16; m >= 1; m >>= 1) s += __shfl_xor(s, m, 32);
  if (lane == 0) out[row] = s + b1[0];
}

__global__ void __launch_bounds__(256)
k_f2bf(const float* __restrict__ src, bf16* __restrict__ dst, int n) {
  const int i = blockIdx.x * 256 + threadIdx.x;
  if (i < n) dst[i] = (bf16)src[i];
}

__global__ void __launch_bounds__(256)
k_zero(float* __restrict__ p, int n) {
  const int i = blockIdx.x * 256 + threadIdx.x;
  if (i < n) p[i] = 0.0f;
}

// ---------------------------------------------------------------------------

extern "C" void kernel_launch(void* const* d_in, const int* in_sizes, int n_in,
                              void* d_out, int out_size, void* d_ws, size_t ws_size,
                              hipStream_t stream) {
  (void)n_in; (void)out_size; (void)ws_size;
  const float* x1d     = (const float*)d_in[0];
  const float* x2d     = (const float*)d_in[1];
  const int*   ei_pipe = (const int*)d_in[2];
  const int*   ei_surf = (const int*)d_in[3];
  const int*   ei_c12  = (const int*)d_in[4];
  const int*   ei_c21  = (const int*)d_in[5];
  const int N1 = in_sizes[0] / 10;
  const int N2 = in_sizes[1] / 15;
  const int Ep = in_sizes[2] / 2, Es = in_sizes[3] / 2;
  const int Ec12 = in_sizes[4] / 2, Ec21 = in_sizes[5] / 2;
  const float* const* P = (const float* const*)(d_in + 6);
  // P[0..3] enc1d, P[4..7] enc2d, P[8..11] dec1d, P[12..15] dec2d,
  // layer l base B=16+16l: pipe{Wl,bl,Wr}=B+0..2, surf=B+3..5,
  // c12=B+6..8, c21=B+9..11, ln1d{g,b}=B+12..13, ln2d{g,b}=B+14..15.

  // --- workspace carve-out (256B aligned) ---
  size_t off = 0;
  auto carve = [&](size_t bytes) -> void* {
    void* p = (char*)d_ws + off;
    off += (bytes + 255) & ~(size_t)255;
    return p;
  };
  bf16*  wbf  = (bf16*)carve((size_t)36 * 16384 * sizeof(bf16));
  float* h1f  = (float*)carve((size_t)N1 * 128 * sizeof(float));
  float* h2f  = (float*)carve((size_t)N2 * 128 * sizeof(float));
  bf16*  h1b  = (bf16*)carve((size_t)N1 * 128 * sizeof(bf16));
  bf16*  h2b  = (bf16*)carve((size_t)N2 * 128 * sizeof(bf16));
  float* new1 = (float*)carve((size_t)N1 * 128 * sizeof(float));
  float* new2 = (float*)carve((size_t)N2 * 128 * sizeof(float));
  float* agg  = (float*)carve((size_t)N2 * 128 * sizeof(float)); // shared scratch
  float* cnt  = (float*)carve((size_t)N2 * sizeof(float));

  // --- bf16 weight slots (36 x 128x128) ---
  const float* wsrc[36];
  wsrc[0] = P[2];  // enc1d[1].W
  wsrc[1] = P[6];  // enc2d[1].W
  wsrc[2] = P[8];  // dec1d[0].W
  wsrc[3] = P[12]; // dec2d[0].W
  for (int l = 0; l < 4; ++l) {
    const int B = 16 + 16 * l, s = 4 + 8 * l;
    wsrc[s + 0] = P[B + 0];  // pipe.Wl
    wsrc[s + 1] = P[B + 2];  // pipe.Wr
    wsrc[s + 2] = P[B + 3];  // surf.Wl
    wsrc[s + 3] = P[B + 5];  // surf.Wr
    wsrc[s + 4] = P[B + 6];  // c12.Wl
    wsrc[s + 5] = P[B + 8];  // c12.Wr
    wsrc[s + 6] = P[B + 9];  // c21.Wl
    wsrc[s + 7] = P[B + 11]; // c21.Wr
  }
  for (int i = 0; i < 36; ++i)
    k_f2bf<<<64, 256, 0, stream>>>(wsrc[i], wbf + (size_t)i * 16384, 16384);

  auto gemm = [&](const bf16* Ab, const float* Af, const float* cp, int slot,
                  const float* bias, float* C, bf16* Cb, int M, int flags) {
    k_wmma_gemm<<<(M + 63) / 64, 256, 0, stream>>>(
        Ab, Af, cp, wbf + (size_t)slot * 16384, bias, C, Cb, M, flags);
  };
  auto zero = [&](float* p, long n) {
    k_zero<<<(int)((n + 255) / 256), 256, 0, stream>>>(p, (int)n);
  };
  auto gather = [&](const bf16* hs, const int* ei, int E) {
    k_gather<<<(int)(((long)E * 32 + 255) / 256), 256, 0, stream>>>(hs, ei, E, agg, cnt);
  };

  // --- encoders ---
  bf16* tmpb = (bf16*)agg; // reuse scratch as bf16 activation buffer
  k_enc1<<<(int)(((long)N1 * 128 + 255) / 256), 256, 0, stream>>>(x1d, P[0], P[1], tmpb, N1, 10);
  gemm(tmpb, nullptr, nullptr, 0, P[3], h1f, h1b, N1, 0);
  k_enc1<<<(int)(((long)N2 * 128 + 255) / 256), 256, 0, stream>>>(x2d, P[4], P[5], tmpb, N2, 15);
  gemm(tmpb, nullptr, nullptr, 1, P[7], h2f, h2b, N2, 0);

  // --- message-passing layers ---
  for (int l = 0; l < 4; ++l) {
    const int B = 16 + 16 * l, s = 4 + 8 * l;
    // dst type 1 (pipes): pipe(h1->h1) + c21(h2->h1)
    zero(agg, (long)N1 * 128); zero(cnt, N1);
    gather(h1b, ei_pipe, Ep);
    gemm(nullptr, agg, cnt, s + 0, P[B + 1],  new1, nullptr, N1, 0);        // mean @ pipe.Wl + bl
    gemm(h1b, nullptr, nullptr, s + 1, nullptr, new1, nullptr, N1, GF_ACC); // + h1 @ pipe.Wr
    zero(agg, (long)N1 * 128); zero(cnt, N1);
    gather(h2b, ei_c21, Ec21);
    gemm(nullptr, agg, cnt, s + 6, P[B + 10], new1, nullptr, N1, GF_ACC);   // + mean @ c21.Wl + bl
    gemm(h1b, nullptr, nullptr, s + 7, nullptr, new1, nullptr, N1, GF_ACC); // + h1 @ c21.Wr
    // dst type 2 (surface): surf(h2->h2) + c12(h1->h2)
    zero(agg, (long)N2 * 128); zero(cnt, N2);
    gather(h2b, ei_surf, Es);
    gemm(nullptr, agg, cnt, s + 2, P[B + 4],  new2, nullptr, N2, 0);
    gemm(h2b, nullptr, nullptr, s + 3, nullptr, new2, nullptr, N2, GF_ACC);
    zero(agg, (long)N2 * 128); zero(cnt, N2);
    gather(h1b, ei_c12, Ec12);
    gemm(nullptr, agg, cnt, s + 4, P[B + 7],  new2, nullptr, N2, GF_ACC);
    gemm(h2b, nullptr, nullptr, s + 5, nullptr, new2, nullptr, N2, GF_ACC);
    // residual + LayerNorm (updates fp32 master and bf16 mirror)
    k_ln<<<(N1 + 7) / 8, 256, 0, stream>>>(h1f, new1, P[B + 12], P[B + 13], h1b, N1);
    k_ln<<<(N2 + 7) / 8, 256, 0, stream>>>(h2f, new2, P[B + 14], P[B + 15], h2b, N2);
  }

  // --- decoders ---
  float* out1 = (float*)d_out;
  float* out2 = out1 + N1;
  gemm(h1b, nullptr, nullptr, 2, P[9],  agg, nullptr, N1, GF_GELU); // hid1 = gelu(h1@W+b)
  k_dec_out<<<(N1 + 7) / 8, 256, 0, stream>>>(agg, P[10], P[11], out1, N1);
  gemm(h2b, nullptr, nullptr, 3, P[13], agg, nullptr, N2, GF_GELU);
  k_dec_out<<<(N2 + 7) / 8, 256, 0, stream>>>(agg, P[14], P[15], out2, N2);
}
